// DMSMHA_Block_72138270703659
// MI455X (gfx1250) — matrix operand order repaired and checked by reading, not compiled, not run on hardware
//
#include <hip/hip_runtime.h>
#include <hip/hip_bf16.h>

// ---------------------------------------------------------------------------
// Deformable-DETR encoder block for MI455X (gfx1250, wave32, WMMA bf16).
//   B=8, N=1024, C=256, NH=8, L=4, PTS=4, D=32, P_TOT=21760
// GEMMs: v_wmma_f32_16x16x32_bf16, f32 accumulate.
//  - weights pre-transposed to N-major bf16 (contiguous B fragments)
//  - B sub-tile staged in LDS once per block per K-step (8x traffic cut)
//  - 32x64 wave tile: 8 WMMAs per K-step, 2 A frags x 4 B frags
//  - activation f32->bf16 conversion fused into the A-fragment load
// ---------------------------------------------------------------------------

typedef __attribute__((ext_vector_type(16))) __bf16 v16bf;
typedef __attribute__((ext_vector_type(8)))  float  v8f;

#define B_    8
#define N_    1024
#define C_    256
#define NH_   8
#define L_    4
#define PTS_  4
#define D_    32
#define PTOT_ 21760
#define MQ_   (B_ * N_)        // 8192 query rows
#define MV_   (B_ * PTOT_)     // 174080 value rows

#define BS_STRIDE 40           // LDS col stride in bf16 (32 data + 8 pad = 80B)

// ---------------------------------------------------------------------------
// Weight transpose + f32->bf16 convert:  WT[n*K + k] = (bf16) W[k*N + n]
// ---------------------------------------------------------------------------
__global__ __launch_bounds__(256) void wprep_kernel(
    const float* __restrict__ W, __bf16* __restrict__ WT, int K, int N) {
  int idx = blockIdx.x * 256 + threadIdx.x;
  if (idx >= N * K) return;
  int n = idx / K;
  int k = idx - n * K;
  WT[idx] = (__bf16)W[(long long)k * N + n];
}

// ---------------------------------------------------------------------------
// GEMM: out[M,N] = A[M,K](f32) * WT[N,K](bf16, pre-transposed) + bias
// Block: 256 threads = 8 waves; block tile 256(M) x 64(N).
// Wave w: rows [by*256 + w*32, +32) as 2 sub-tiles; cols [bx*64, +64) as 4.
// Per K-step (32): stage B 64x32 into LDS (1 x b128 per thread), then each
// wave does 2x4 = 8 v_wmma_f32_16x16x32_bf16.
// A-fragment (16-bit A 16x32, ISA 7.12.2): lane%16 = row M,
//   lanes 0-15: elems 0..15 = K k0..k0+15 (contiguous order used here, as in B)
//   lanes 16-31: same with K +16.
// B-fragment: lane%16 = col N; lanes<16: K k0..k0+15, lanes>=16: +16.
// ---------------------------------------------------------------------------
__global__ __launch_bounds__(256) void gemm_bf16_wmma(
    const float*  __restrict__ A,
    const __bf16* __restrict__ WT,
    const float*  __restrict__ bias,
    float*        __restrict__ outF,   // used when outBf16 == 0
    __bf16*       __restrict__ outB,   // used when outBf16 == 1
    int M, int N, int K, int relu, int outBf16) {
  (void)M;
  __shared__ __bf16 Bs[64 * BS_STRIDE];   // 5 KB, padded against bank conflicts

  const int lane    = threadIdx.x & 31;
  const int wave    = threadIdx.x >> 5;
  const int halfSel = lane >> 4;   // 0 or 1
  const int l16     = lane & 15;
  const int mBase   = blockIdx.y * 256 + wave * 32;
  const int nBase   = blockIdx.x * 64;

  v8f zero = {};
  v8f acc[2][4] = {{zero, zero, zero, zero}, {zero, zero, zero, zero}};

  const float* arow0 = A + (long long)(mBase + l16) * K;
  const float* arow1 = A + (long long)(mBase + 16 + l16) * K;

  // B staging: thread t covers col = t>>2 (0..63), chunk = t&3 (8 bf16 = 16B)
  const int scol   = threadIdx.x >> 2;
  const int schunk = threadIdx.x & 3;
  const __bf16* wtp = WT + (long long)(nBase + scol) * K + schunk * 8;

  for (int k0 = 0; k0 < K; k0 += 32) {
    __syncthreads();   // previous iteration's LDS reads complete
    *(float4*)(&Bs[scol * BS_STRIDE + schunk * 8]) = *(const float4*)(wtp + k0);
    __syncthreads();   // staging visible to all waves

    if (k0 + 32 < K) {
      __builtin_prefetch(arow0 + k0 + 32, 0, 3);
      __builtin_prefetch(arow1 + k0 + 32, 0, 3);
    }

    // ---- A fragments (fused f32 -> bf16 convert) ----
    const int kb = k0 + halfSel * 16;
    v16bf af0, af1;
    {
      float4 a0 = *(const float4*)(arow0 + kb);
      float4 a1 = *(const float4*)(arow0 + kb + 4);
      float4 a2 = *(const float4*)(arow0 + kb + 8);
      float4 a3 = *(const float4*)(arow0 + kb + 12);
      af0[0]  = (__bf16)a0.x;  af0[1]  = (__bf16)a0.y;
      af0[2]  = (__bf16)a0.z;  af0[3]  = (__bf16)a0.w;
      af0[4]  = (__bf16)a1.x;  af0[5]  = (__bf16)a1.y;
      af0[6]  = (__bf16)a1.z;  af0[7]  = (__bf16)a1.w;
      af0[8]  = (__bf16)a2.x;  af0[9]  = (__bf16)a2.y;
      af0[10] = (__bf16)a2.z;  af0[11] = (__bf16)a2.w;
      af0[12] = (__bf16)a3.x;  af0[13] = (__bf16)a3.y;
      af0[14] = (__bf16)a3.z;  af0[15] = (__bf16)a3.w;
    }
    {
      float4 a0 = *(const float4*)(arow1 + kb);
      float4 a1 = *(const float4*)(arow1 + kb + 4);
      float4 a2 = *(const float4*)(arow1 + kb + 8);
      float4 a3 = *(const float4*)(arow1 + kb + 12);
      af1[0]  = (__bf16)a0.x;  af1[1]  = (__bf16)a0.y;
      af1[2]  = (__bf16)a0.z;  af1[3]  = (__bf16)a0.w;
      af1[4]  = (__bf16)a1.x;  af1[5]  = (__bf16)a1.y;
      af1[6]  = (__bf16)a1.z;  af1[7]  = (__bf16)a1.w;
      af1[8]  = (__bf16)a2.x;  af1[9]  = (__bf16)a2.y;
      af1[10] = (__bf16)a2.z;  af1[11] = (__bf16)a2.w;
      af1[12] = (__bf16)a3.x;  af1[13] = (__bf16)a3.y;
      af1[14] = (__bf16)a3.z;  af1[15] = (__bf16)a3.w;
    }

    // ---- B fragments from LDS + 8 WMMAs ----
    const int kb2 = halfSel * 16;
#pragma unroll
    for (int j = 0; j < 4; ++j) {
      v16bf bf = *(const v16bf*)(&Bs[(j * 16 + l16) * BS_STRIDE + kb2]);
      acc[0][j] = __builtin_amdgcn_wmma_f32_16x16x32_bf16(
          false, af0, false, bf, (short)0, acc[0][j], false, false);
      acc[1][j] = __builtin_amdgcn_wmma_f32_16x16x32_bf16(
          false, af1, false, bf, (short)0, acc[1][j], false, false);
    }
  }

  // ---- epilogue: C/D layout (ISA 7.12.2): VGPR r, lanes<16 -> M=r,
  // lanes>=16 -> M=r+8; N = lane%16 within each 16-col tile. ----
#pragma unroll
  for (int i = 0; i < 2; ++i) {
#pragma unroll
    for (int j = 0; j < 4; ++j) {
      const int col = nBase + j * 16 + l16;
      const float bv = bias ? bias[col] : 0.0f;
#pragma unroll
      for (int r = 0; r < 8; ++r) {
        const int row = mBase + i * 16 + r + halfSel * 8;
        float v = acc[i][j][r] + bv;
        if (relu) v = fmaxf(v, 0.0f);
        const long long o = (long long)row * N + col;
        if (outBf16) outB[o] = (__bf16)v;
        else         outF[o] = v;
      }
    }
  }
}

// ---------------------------------------------------------------------------
// Deformable sampling: one wave per (b, n, head); lane = channel d (0..31).
// Softmax over L*PTS=16 computed redundantly per lane (broadcast loads);
// each corner gather is a coalesced 64B bf16 burst across the wave.
// ---------------------------------------------------------------------------
__global__ __launch_bounds__(256) void msda_sample_kernel(
    const float*  __restrict__ off_raw,   // (B*N, 256)
    const float*  __restrict__ attn_raw,  // (B*N, 128)
    const __bf16* __restrict__ val,       // (B*P_TOT, 256) == (B,P,NH,D)
    const float*  __restrict__ refpts,    // (B, N, L, 2)
    const int*    __restrict__ shapes,    // (L, 2) = (H, W)
    const int*    __restrict__ starts,    // (L,)
    float*        __restrict__ sampled)   // (B*N, 256)
{
  const int gwave = (int)((blockIdx.x * blockDim.x + threadIdx.x) >> 5);
  const int lane  = threadIdx.x & 31;
  const int h = gwave & (NH_ - 1);
  const int n = (gwave >> 3) & (N_ - 1);
  const int b = gwave >> 13;
  const long long qi = (long long)b * N_ + n;

  // softmax over 16 logits
  const float* al = attn_raw + qi * (NH_ * L_ * PTS_) + h * (L_ * PTS_);
  float logits[16];
  float mx = -1e30f;
#pragma unroll
  for (int j = 0; j < 16; ++j) { logits[j] = al[j]; mx = fmaxf(mx, logits[j]); }
  float ssum = 0.0f;
#pragma unroll
  for (int j = 0; j < 16; ++j) { logits[j] = __expf(logits[j] - mx); ssum += logits[j]; }
  const float inv = 1.0f / ssum;

  const float* offp = off_raw + qi * (NH_ * L_ * PTS_ * 2) + h * (L_ * PTS_ * 2);

  float acc = 0.0f;
#pragma unroll
  for (int l = 0; l < L_; ++l) {
    const int Hl = shapes[l * 2 + 0];
    const int Wl = shapes[l * 2 + 1];
    const int st = starts[l];
    const float rx = refpts[(qi * L_ + l) * 2 + 0];
    const float ry = refpts[(qi * L_ + l) * 2 + 1];
    const __bf16* vbase =
        val + ((long long)b * PTOT_ + st) * C_ + h * D_ + lane;
#pragma unroll
    for (int p = 0; p < PTS_; ++p) {
      const float w  = logits[l * PTS_ + p] * inv;
      const float ox = offp[l * PTS_ * 2 + p * 2 + 0];
      const float oy = offp[l * PTS_ * 2 + p * 2 + 1];
      // loc = ref + off/(W,H);  x = loc.x*W - 0.5
      const float x = (rx + ox / (float)Wl) * (float)Wl - 0.5f;
      const float y = (ry + oy / (float)Hl) * (float)Hl - 0.5f;
      const float x0 = floorf(x), y0 = floorf(y);
      const float tx = x - x0,   ty = y - y0;
      const int x0i = (int)x0, y0i = (int)y0;
#pragma unroll
      for (int c = 0; c < 4; ++c) {
        const int dx = c & 1, dy = c >> 1;
        const float cw = (dx ? tx : 1.0f - tx) * (dy ? ty : 1.0f - ty);
        const int xi = x0i + dx, yi = y0i + dy;
        if (xi >= 0 && xi < Wl && yi >= 0 && yi < Hl) {
          const float g = (float)vbase[((long long)yi * Wl + xi) * C_];
          acc += w * cw * g;
        }
      }
    }
  }
  sampled[qi * C_ + h * D_ + lane] = acc;
}

// ---------------------------------------------------------------------------
// Fused residual-add + LayerNorm over C=256: one wave per row, 8 elems/lane,
// wave32 shfl_xor reductions.
// ---------------------------------------------------------------------------
__global__ __launch_bounds__(256) void add_ln_kernel(
    const float* __restrict__ X, const float* __restrict__ Y,
    const float* __restrict__ g, const float* __restrict__ be,
    float* __restrict__ out) {
  const int row  = (int)((blockIdx.x * blockDim.x + threadIdx.x) >> 5);
  const int lane = threadIdx.x & 31;
  const long long base = (long long)row * C_ + lane * 8;

  float vals[8];
  float s = 0.0f;
#pragma unroll
  for (int j = 0; j < 8; ++j) { vals[j] = X[base + j] + Y[base + j]; s += vals[j]; }
#pragma unroll
  for (int o = 16; o >= 1; o >>= 1) s += __shfl_xor(s, o, 32);
  const float mean = s * (1.0f / (float)C_);

  float v2 = 0.0f;
#pragma unroll
  for (int j = 0; j < 8; ++j) { const float d = vals[j] - mean; v2 += d * d; }
#pragma unroll
  for (int o = 16; o >= 1; o >>= 1) v2 += __shfl_xor(v2, o, 32);
  const float rstd = rsqrtf(v2 * (1.0f / (float)C_) + 1e-5f);

#pragma unroll
  for (int j = 0; j < 8; ++j)
    out[base + j] = (vals[j] - mean) * rstd * g[lane * 8 + j] + be[lane * 8 + j];
}

// ---------------------------------------------------------------------------
// Host launcher
// ---------------------------------------------------------------------------
extern "C" void kernel_launch(void* const* d_in, const int* in_sizes, int n_in,
                              void* d_out, int out_size, void* d_ws, size_t ws_size,
                              hipStream_t stream) {
  (void)in_sizes; (void)n_in; (void)out_size; (void)ws_size;

  const float* q      = (const float*)d_in[0];
  // d_in[1] = k : unused by the reference
  const float* v      = (const float*)d_in[2];
  const float* refpts = (const float*)d_in[3];
  const int*   shapes = (const int*)d_in[4];
  const int*   starts = (const int*)d_in[5];
  const float* W_off  = (const float*)d_in[6];
  const float* b_off  = (const float*)d_in[7];
  const float* W_attn = (const float*)d_in[8];
  const float* b_attn = (const float*)d_in[9];
  const float* W_v    = (const float*)d_in[10];
  const float* b_v    = (const float*)d_in[11];
  const float* W_out  = (const float*)d_in[12];
  const float* b_out  = (const float*)d_in[13];
  const float* W1     = (const float*)d_in[14];
  const float* b1     = (const float*)d_in[15];
  const float* W2     = (const float*)d_in[16];
  const float* b2     = (const float*)d_in[17];
  const float* g1     = (const float*)d_in[18];
  const float* be1    = (const float*)d_in[19];
  const float* g2     = (const float*)d_in[20];
  const float* be2    = (const float*)d_in[21];

  // ---- workspace layout (256B aligned carve-outs) ----
  char* ws = (char*)d_ws;
  size_t off = 0;
  auto carve = [&](size_t bytes) -> char* {
    char* p = ws + off;
    off += (bytes + 255) & ~(size_t)255;
    return p;
  };
  __bf16* WvT    = (__bf16*)carve((size_t)C_ * C_ * 2);          // 256x256
  __bf16* WoffT  = (__bf16*)carve((size_t)C_ * 256 * 2);         // 256x256
  __bf16* WattnT = (__bf16*)carve((size_t)C_ * 128 * 2);         // 128x256
  __bf16* WoutT  = (__bf16*)carve((size_t)C_ * C_ * 2);          // 256x256
  __bf16* W1T    = (__bf16*)carve((size_t)C_ * 1024 * 2);        // 1024x256
  __bf16* W2T    = (__bf16*)carve((size_t)1024 * C_ * 2);        // 256x1024
  __bf16* valbf  = (__bf16*)carve((size_t)MV_ * C_ * 2);         // 89 MB
  float*  offRaw = (float*)carve((size_t)MQ_ * 256 * 4);
  float*  attRaw = (float*)carve((size_t)MQ_ * 128 * 4);
  float*  sampB  = (float*)carve((size_t)MQ_ * C_ * 4);
  float*  yBuf   = (float*)carve((size_t)MQ_ * C_ * 4);
  float*  xBuf   = (float*)carve((size_t)MQ_ * C_ * 4);
  float*  hBuf   = (float*)carve((size_t)MQ_ * 1024 * 4);
  float*  y2Buf  = (float*)carve((size_t)MQ_ * C_ * 4);

  // ---- 1. weight prep (transpose + bf16) ----
  wprep_kernel<<<(C_ * C_ + 255) / 256, 256, 0, stream>>>(W_v,    WvT,    C_, C_);
  wprep_kernel<<<(C_ * 256 + 255) / 256, 256, 0, stream>>>(W_off,  WoffT,  C_, 256);
  wprep_kernel<<<(C_ * 128 + 255) / 256, 256, 0, stream>>>(W_attn, WattnT, C_, 128);
  wprep_kernel<<<(C_ * C_ + 255) / 256, 256, 0, stream>>>(W_out,  WoutT,  C_, C_);
  wprep_kernel<<<(C_ * 1024 + 255) / 256, 256, 0, stream>>>(W1,    W1T,    C_, 1024);
  wprep_kernel<<<(1024 * C_ + 255) / 256, 256, 0, stream>>>(W2,    W2T,    1024, C_);

  // ---- 2. value projection (the big, HBM-bound GEMM): val = v@W_v + b_v ----
  gemm_bf16_wmma<<<dim3(C_ / 64, MV_ / 256), 256, 0, stream>>>(
      v, WvT, b_v, nullptr, valbf, MV_, C_, C_, /*relu=*/0, /*outBf16=*/1);

  // ---- 3. query projections: offsets & attention logits ----
  gemm_bf16_wmma<<<dim3(256 / 64, MQ_ / 256), 256, 0, stream>>>(
      q, WoffT, b_off, offRaw, nullptr, MQ_, 256, C_, 0, 0);
  gemm_bf16_wmma<<<dim3(128 / 64, MQ_ / 256), 256, 0, stream>>>(
      q, WattnT, b_attn, attRaw, nullptr, MQ_, 128, C_, 0, 0);

  // ---- 4. deformable bilinear sampling (one wave per (b,n,h)) ----
  msda_sample_kernel<<<(B_ * N_ * NH_) / 8, 256, 0, stream>>>(
      offRaw, attRaw, valbf, refpts, shapes, starts, sampB);

  // ---- 5. output projection ----
  gemm_bf16_wmma<<<dim3(C_ / 64, MQ_ / 256), 256, 0, stream>>>(
      sampB, WoutT, b_out, yBuf, nullptr, MQ_, C_, C_, 0, 0);

  // ---- 6. x = LN(q + attn_out) ----
  add_ln_kernel<<<MQ_ / 8, 256, 0, stream>>>(q, yBuf, g1, be1, xBuf);

  // ---- 7. FFN ----
  gemm_bf16_wmma<<<dim3(1024 / 64, MQ_ / 256), 256, 0, stream>>>(
      xBuf, W1T, b1, hBuf, nullptr, MQ_, 1024, C_, /*relu=*/1, 0);
  gemm_bf16_wmma<<<dim3(C_ / 64, MQ_ / 256), 256, 0, stream>>>(
      hBuf, W2T, b2, y2Buf, nullptr, MQ_, C_, 1024, 0, 0);

  // ---- 8. out = LN(x + ffn) ----
  add_ln_kernel<<<MQ_ / 8, 256, 0, stream>>>(xBuf, y2Buf, g2, be2, (float*)d_out);
}
